// ModelBasedNet_19499151524109
// MI455X (gfx1250) — matrix operand order repaired and checked by reading, not compiled, not run on hardware
//
#include <hip/hip_runtime.h>
#include <hip/hip_bf16.h>

// ---------------------------------------------------------------------------
// Problem constants (from reference)
// ---------------------------------------------------------------------------
#define BSZ   512
#define NF    128
#define NA    200
#define HID   256
#define NAPAD 208          // NA padded to 13 16x16 blocks
#define NNEWT 20
#define NB    16           // Cholesky block size (WMMA tile)
#define NBLK  13           // 208/16
#define LD    212          // LDS row stride: 16B-aligned rows (async B128), 16-bank spread

typedef __attribute__((ext_vector_type(16))) __bf16 v16bf;
typedef __attribute__((ext_vector_type(8)))  float  v8f;
typedef __attribute__((ext_vector_type(2)))  float  v2f;

union Frag { v16bf v; unsigned int u[8]; };

// ---------------------------------------------------------------------------
// fp32 -> bf16 conversion kernels
// ---------------------------------------------------------------------------
__global__ void cvt_bf16_kernel(const float* __restrict__ src,
                                __hip_bfloat16* __restrict__ dst, int n) {
    int i = blockIdx.x * blockDim.x + threadIdx.x;
    if (i < n) dst[i] = __float2bfloat16(src[i]);
}

__global__ void cvt_w2_pad_kernel(const float* __restrict__ W2,
                                  __hip_bfloat16* __restrict__ dst) {
    int i = blockIdx.x * blockDim.x + threadIdx.x;
    if (i < NAPAD * HID) {
        int r = i / HID, c = i % HID;
        float v = (r < NA) ? W2[r * HID + c] : 0.0f;
        dst[i] = __float2bfloat16(v);
    }
}

// ---------------------------------------------------------------------------
// GEMM1: h = LeakyReLU(x @ W1^T + b1)  -> bf16 (512x256), WMMA bf16
// ---------------------------------------------------------------------------
__global__ __launch_bounds__(256) void gemm1_kernel(
    const __hip_bfloat16* __restrict__ xb,
    const __hip_bfloat16* __restrict__ w1b,
    const float* __restrict__ b1,
    __hip_bfloat16* __restrict__ hb)
{
    const int lane = threadIdx.x & 31;
    const int wave = (blockIdx.x * blockDim.x + threadIdx.x) >> 5;
    const int NTN  = HID / 16;
    const int tm = wave / NTN, tn = wave % NTN;

    const int mrow = tm * 16 + (lane & 15);
    const int ncol = tn * 16 + (lane & 15);
    const int khA  = (lane < 16) ? 0 : 8;
    const int khB  = (lane < 16) ? 0 : 8;

    v8f acc = {};
    for (int kk = 0; kk < NF; kk += 32) {
        Frag a, b;
#pragma unroll
        for (int v = 0; v < 8; ++v) {
            int ka = kk + ((v < 4) ? 0 : 16) + khA + 2 * (v & 3);
            a.u[v] = *(const unsigned int*)(xb + mrow * NF + ka);
            int kb = kk + ((v < 4) ? 0 : 16) + khB + 2 * (v & 3);
            b.u[v] = *(const unsigned int*)(w1b + ncol * NF + kb);
        }
        acc = __builtin_amdgcn_wmma_f32_16x16x32_bf16(
            false, a.v, false, b.v, (short)0, acc, false, false);
    }

    const int mbase = tm * 16 + ((lane < 16) ? 0 : 8);
    const float bias = b1[ncol];
#pragma unroll
    for (int r = 0; r < 8; ++r) {
        int m = mbase + r;
        float val = acc[r] + bias;
        val = (val >= 0.0f) ? val : 0.01f * val;
        hb[m * HID + ncol] = __float2bfloat16(val);
    }
}

// ---------------------------------------------------------------------------
// GEMM2: logits = h @ W2^T + b2 -> f32 (512x208), WMMA bf16
// ---------------------------------------------------------------------------
__global__ __launch_bounds__(256) void gemm2_kernel(
    const __hip_bfloat16* __restrict__ hb,
    const __hip_bfloat16* __restrict__ w2b,
    const float* __restrict__ b2,
    float* __restrict__ logits)
{
    const int lane = threadIdx.x & 31;
    const int wave = (blockIdx.x * blockDim.x + threadIdx.x) >> 5;
    const int NTN  = NAPAD / 16;
    const int tm = wave / NTN, tn = wave % NTN;

    const int mrow = tm * 16 + (lane & 15);
    const int ncol = tn * 16 + (lane & 15);
    const int khA  = (lane < 16) ? 0 : 8;
    const int khB  = (lane < 16) ? 0 : 8;

    v8f acc = {};
    for (int kk = 0; kk < HID; kk += 32) {
        Frag a, b;
#pragma unroll
        for (int v = 0; v < 8; ++v) {
            int ka = kk + ((v < 4) ? 0 : 16) + khA + 2 * (v & 3);
            a.u[v] = *(const unsigned int*)(hb + mrow * HID + ka);
            int kb = kk + ((v < 4) ? 0 : 16) + khB + 2 * (v & 3);
            b.u[v] = *(const unsigned int*)(w2b + ncol * HID + kb);
        }
        acc = __builtin_amdgcn_wmma_f32_16x16x32_bf16(
            false, a.v, false, b.v, (short)0, acc, false, false);
    }

    const int mbase = tm * 16 + ((lane < 16) ? 0 : 8);
    const float bias = (ncol < NA) ? b2[ncol] : 0.0f;
#pragma unroll
    for (int r = 0; r < 8; ++r) {
        int m = mbase + r;
        logits[m * NAPAD + ncol] = acc[r] + bias;
    }
}

// ---------------------------------------------------------------------------
// Softmax + clamp + renormalize. One block per row.
// ---------------------------------------------------------------------------
__global__ __launch_bounds__(256) void softmax_kernel(
    const float* __restrict__ logits,
    float* __restrict__ b_out,
    float* __restrict__ bc)
{
    __shared__ float red[256];
    const int m = blockIdx.x;
    const int t = threadIdx.x;

    float v = (t < NA) ? logits[m * NAPAD + t] : -3.0e38f;
    red[t] = v;
    __syncthreads();
    for (int s = 128; s > 0; s >>= 1) {
        if (t < s) red[t] = fmaxf(red[t], red[t + s]);
        __syncthreads();
    }
    float mx = red[0];
    __syncthreads();

    float e = (t < NA) ? __expf(v - mx) : 0.0f;
    red[t] = e;
    __syncthreads();
    for (int s = 128; s > 0; s >>= 1) {
        if (t < s) red[t] += red[t + s];
        __syncthreads();
    }
    float inv_sum = 1.0f / red[0];
    __syncthreads();

    float bsm = e * inv_sum;
    if (t < NA) b_out[m * NA + t] = bsm;

    float c = (t < NA) ? fmaxf(bsm, 0.0001f) : 0.0f;
    red[t] = c;
    __syncthreads();
    for (int s = 128; s > 0; s >>= 1) {
        if (t < s) red[t] += red[t + s];
        __syncthreads();
    }
    float inv_s2 = 1.0f / red[0];
    if (t < NA) bc[m * NA + t] = c * inv_s2;
}

// ---------------------------------------------------------------------------
// Per-sample damped-Newton risk-budget solve. One block (256 thr) per sample.
// Sigma staged to LDS via GLOBAL_LOAD_ASYNC_TO_LDS_B128 (ASYNCcnt).
// Blocked Cholesky (NB=16) with trailing updates on v_wmma_f32_16x16x4_f32.
// Matrix padded to 208x208 (identity pad block) so all 13 blocks are uniform.
// ---------------------------------------------------------------------------
__global__ __launch_bounds__(256) void newton_kernel(
    const float* __restrict__ Sigma,
    const float* __restrict__ bc,
    float* __restrict__ zout)
{
    __shared__ __align__(16) float Hs[NAPAD * LD];
    __shared__ float yv[NAPAD], bv[NAPAD], rhs[NAPAD], wv[NAPAD], dv[NAPAD];
    __shared__ float red[256];

    const int t = threadIdx.x;
    const int lane = t & 31;
    const int wave = t >> 5;
    const int sample = blockIdx.x;
    if (t < NA) bv[t] = bc[sample * NA + t];
    const float4* Sg = (const float4*)(Sigma + (size_t)sample * NA * NA);

    for (int iter = 0; iter < NNEWT; ++iter) {
        __syncthreads();
        // ---- stage Sigma (200x200) into padded LDS via async B128 copies ----
        for (int idx = t; idx < NA * (NA / 4); idx += 256) {
            int r = idx / (NA / 4), c = (idx % (NA / 4)) * 4;
            unsigned lds = (unsigned)(unsigned long long)(const void*)&Hs[r * LD + c];
            unsigned long long ga = (unsigned long long)(const void*)(Sg + idx);
            asm volatile("global_load_async_to_lds_b128 %0, %1, off"
                         :: "v"(lds), "v"(ga) : "memory");
        }
        // pad cols 200..207 of real rows: zero
        for (int idx = t; idx < NA * 8; idx += 256) {
            int r = idx >> 3, c = NA + (idx & 7);
            Hs[r * LD + c] = 0.0f;
        }
        // pad rows 200..207: identity
        for (int idx = t; idx < 8 * NAPAD; idx += 256) {
            int r = NA + idx / NAPAD, c = idx % NAPAD;
            Hs[r * LD + c] = (r == c) ? 1.0f : 0.0f;
        }
        if (t >= NA && t < NAPAD) rhs[t] = 0.0f;
        asm volatile("s_wait_asynccnt 0x0" ::: "memory");
        __syncthreads();

        if (iter == 0) {
            // y0 = b / sqrt(b^T S b)
            float p = 0.0f;
            if (t < NA) {
                float a = 0.0f;
                for (int j = 0; j < NA; ++j) a += Hs[t * LD + j] * bv[j];
                p = bv[t] * a;
            }
            red[t] = p;
            __syncthreads();
            for (int s = 128; s > 0; s >>= 1) {
                if (t < s) red[t] += red[t + s];
                __syncthreads();
            }
            float s0 = red[0];
            if (t < NA) yv[t] = bv[t] * rsqrtf(s0);
            __syncthreads();
        }

        // ---- H = S + diag(b/y^2) ----
        if (t < NA) {
            float yi = yv[t];
            Hs[t * LD + t] += bv[t] / (yi * yi);
        }
        __syncthreads();

        // ---- rhs = -g = -(H y - 2 b/y) ----
        if (t < NA) {
            float a = 0.0f;
            for (int j = 0; j < NA; ++j) a += Hs[t * LD + j] * yv[j];
            rhs[t] = -(a - 2.0f * bv[t] / yv[t]);
        }
        __syncthreads();

        // ================= blocked Cholesky, 13 block-columns =================
        for (int kb = 0; kb < NBLK; ++kb) {
            const int o = kb * NB;

            // (a) 16x16 diagonal factor by wave 0 (lockstep, in-order LDS)
            if (wave == 0) {
                volatile float* vH = Hs;
                const int i = lane;
                for (int k = 0; k < NB; ++k) {
                    float dkk = vH[(o + k) * LD + o + k];
                    float l = sqrtf(dkk);
                    float invl = 1.0f / l;
                    if (i == k) vH[(o + k) * LD + o + k] = l;
                    if (i > k && i < NB) vH[(o + i) * LD + o + k] *= invl;
                    __builtin_amdgcn_wave_barrier();
                    if (i > k && i < NB) {
                        float lik = vH[(o + i) * LD + o + k];
                        for (int j = k + 1; j <= i; ++j)
                            vH[(o + i) * LD + o + j] -= lik * vH[(o + j) * LD + o + k];
                    }
                    __builtin_amdgcn_wave_barrier();
                }
            }
            __syncthreads();

            // (b) panel solve: X * D^T = A for all rows below the block
            const int nrows = (NBLK - 1 - kb) * NB;
            if (t < nrows) {
                int gi = o + NB + t;
                float x[NB];
#pragma unroll
                for (int c = 0; c < NB; ++c) {
                    float a = Hs[gi * LD + o + c];
                    for (int j = 0; j < c; ++j) a -= x[j] * Hs[(o + c) * LD + o + j];
                    x[c] = a / Hs[(o + c) * LD + o + c];
                    Hs[gi * LD + o + c] = x[c];
                }
            }
            __syncthreads();

            // (c) trailing update A[ib][jb] -= L[ib][kb] * L[jb][kb]^T via
            //     4x v_wmma_f32_16x16x4_f32 (K=16). One tile per wave.
            const int tcnt = (NBLK - 1 - kb) * (NBLK - kb) / 2;
            for (int tile = wave; tile < tcnt; tile += 8) {
                int rem = tile, jb = kb + 1, len = NBLK - jb;
                while (rem >= len) { rem -= len; ++jb; --len; }
                int ib = jb + rem;
                int io = ib * NB, jo = jb * NB;

                const int m  = lane & 15;
                const int kh = (lane < 16) ? 0 : 2;
                v8f acc = {};
#pragma unroll
                for (int k4 = 0; k4 < 4; ++k4) {
                    int kbase = o + k4 * 4 + kh;
                    v2f af, bf;
                    af[0] = Hs[(io + m) * LD + kbase];
                    af[1] = Hs[(io + m) * LD + kbase + 1];
                    bf[0] = Hs[(jo + m) * LD + kbase];
                    bf[1] = Hs[(jo + m) * LD + kbase + 1];
                    acc = __builtin_amdgcn_wmma_f32_16x16x4_f32(
                        false, af, false, bf, (short)0, acc, false, false);
                }
                const int mb = (lane < 16) ? 0 : 8;
                const int n  = lane & 15;
#pragma unroll
                for (int r = 0; r < 8; ++r)
                    Hs[(io + mb + r) * LD + jo + n] -= acc[r];
            }
            __syncthreads();
        }

        // ================= blocked forward solve: L w = rhs =================
        for (int kb = 0; kb < NBLK; ++kb) {
            const int o = kb * NB;
            if (t == 0) {
                for (int k = 0; k < NB; ++k) {
                    float a = rhs[o + k];
                    for (int j = 0; j < k; ++j) a -= Hs[(o + k) * LD + o + j] * wv[o + j];
                    wv[o + k] = a / Hs[(o + k) * LD + o + k];
                }
            }
            __syncthreads();
            const int nrows = NAPAD - (o + NB);
            if (t < nrows) {
                int gi = o + NB + t;
                float a = rhs[gi];
#pragma unroll
                for (int j = 0; j < NB; ++j) a -= Hs[gi * LD + o + j] * wv[o + j];
                rhs[gi] = a;
            }
            __syncthreads();
        }

        // ================= blocked back solve: L^T d = w =================
        for (int kb = NBLK - 1; kb >= 0; --kb) {
            const int o = kb * NB;
            if (t == 0) {
                for (int k = NB - 1; k >= 0; --k) {
                    float a = wv[o + k];
                    for (int j = k + 1; j < NB; ++j) a -= Hs[(o + j) * LD + o + k] * dv[o + j];
                    dv[o + k] = a / Hs[(o + k) * LD + o + k];
                }
            }
            __syncthreads();
            if (t < o) {
                float a = wv[t];
#pragma unroll
                for (int j = 0; j < NB; ++j) a -= Hs[(o + j) * LD + t] * dv[o + j];
                wv[t] = a;
            }
            __syncthreads();
        }

        // ---- damped step keeping y > 0 ----
        float loc = 1.0e30f;
        if (t < NA) {
            float d = dv[t];
            if (d < 0.0f) loc = -yv[t] / d;
        }
        red[t] = loc;
        __syncthreads();
        for (int s = 128; s > 0; s >>= 1) {
            if (t < s) red[t] = fminf(red[t], red[t + s]);
            __syncthreads();
        }
        float tstep = fminf(1.0f, 0.9f * red[0]);
        __syncthreads();
        if (t < NA) yv[t] = fmaxf(yv[t] + tstep * dv[t], 1e-12f);
    }
    __syncthreads();

    // ---- z = y / sum(y) ----
    red[t] = (t < NA) ? yv[t] : 0.0f;
    __syncthreads();
    for (int s = 128; s > 0; s >>= 1) {
        if (t < s) red[t] += red[t + s];
        __syncthreads();
    }
    float inv = 1.0f / red[0];
    if (t < NA) zout[sample * NA + t] = yv[t] * inv;
}

// ---------------------------------------------------------------------------
// Launcher
// ---------------------------------------------------------------------------
extern "C" void kernel_launch(void* const* d_in, const int* in_sizes, int n_in,
                              void* d_out, int out_size, void* d_ws, size_t ws_size,
                              hipStream_t stream) {
    const float* x     = (const float*)d_in[0];
    const float* Sigma = (const float*)d_in[1];
    const float* W1    = (const float*)d_in[2];
    const float* b1    = (const float*)d_in[3];
    const float* W2    = (const float*)d_in[4];
    const float* b2    = (const float*)d_in[5];

    float* zout = (float*)d_out;
    float* bout = zout + BSZ * NA;

    char* ws = (char*)d_ws;
    __hip_bfloat16* xb   = (__hip_bfloat16*)(ws + 0);
    __hip_bfloat16* w1b  = (__hip_bfloat16*)(ws + 131072);
    __hip_bfloat16* hb   = (__hip_bfloat16*)(ws + 196608);
    __hip_bfloat16* w2b  = (__hip_bfloat16*)(ws + 458752);
    float*          logi = (float*)(ws + 565248);
    float*          bcw  = (float*)(ws + 991232);

    cvt_bf16_kernel<<<(BSZ * NF + 255) / 256, 256, 0, stream>>>(x, xb, BSZ * NF);
    cvt_bf16_kernel<<<(HID * NF + 255) / 256, 256, 0, stream>>>(W1, w1b, HID * NF);
    cvt_w2_pad_kernel<<<(NAPAD * HID + 255) / 256, 256, 0, stream>>>(W2, w2b);

    gemm1_kernel<<<(BSZ / 16) * (HID / 16) * 32 / 256, 256, 0, stream>>>(xb, w1b, b1, hb);
    gemm2_kernel<<<(BSZ / 16) * (NAPAD / 16) * 32 / 256, 256, 0, stream>>>(hb, w2b, b2, logi);

    softmax_kernel<<<BSZ, 256, 0, stream>>>(logi, bout, bcw);

    newton_kernel<<<BSZ, 256, 0, stream>>>(Sigma, bcw, zout);
}
